// GINEncoder_74749610819666
// MI455X (gfx1250) — matrix operand order repaired and checked by reading, not compiled, not run on hardware
//
#include <hip/hip_runtime.h>

typedef __attribute__((ext_vector_type(16))) __bf16 v16bf;
typedef __attribute__((ext_vector_type(8)))  float  v8f;

#define N_NODES  100000
#define N_EDGES  600000
#define N_GRAPHS 1024
#define F_IN     32
#define GHC      128
#define NHID     512
#define NOUT     256
#define N_TILES  (N_NODES / 16)   // 6250, exact
#define BN_EPS   1e-5f

// ---------------------------------------------------------------- utilities
__global__ void zero_f32(float* __restrict__ p, int n) {
    int i = blockIdx.x * blockDim.x + threadIdx.x;
    if (i < n) p[i] = 0.0f;
}

__global__ void copy_f32(float* __restrict__ d, const float* __restrict__ s, int n) {
    int i = blockIdx.x * blockDim.x + threadIdx.x;
    if (i < n) d[i] = s[i];
}

__global__ void count_nodes(const int* __restrict__ batch, float* __restrict__ cnt) {
    int i = blockIdx.x * blockDim.x + threadIdx.x;
    if (i < N_NODES) unsafeAtomicAdd(&cnt[batch[i]], 1.0f);
}

// Pack a [K][N] f32 row-major weight into the WMMA bf16 B-fragment order:
// idx = ((nt*(K/32)+kc)*32 + lane)*16 + j ;  n = nt*16 + (lane&15) ;
// k = kc*32 + 16*(j>>3) + 8*(lane>>4) + (j&7)
__global__ void pack_bfrags(const float* __restrict__ W, __bf16* __restrict__ out,
                            int K, int N) {
    int idx = blockIdx.x * blockDim.x + threadIdx.x;
    if (idx >= K * N) return;
    int j    = idx & 15;
    int lane = (idx >> 4) & 31;
    int rest = idx >> 9;
    int kcn  = K >> 5;
    int kc   = rest % kcn;
    int nt   = rest / kcn;
    int n = nt * 16 + (lane & 15);
    int k = kc * 32 + ((j >> 3) << 4) + ((lane >> 4) << 3) + (j & 7);
    out[idx] = (__bf16)W[(size_t)k * N + n];
}

// -------------------------------------------------- edge gather/scatter-add
template <int F4>   // F4 = F/4 (8 for layer 0, 32 otherwise)
__global__ void edge_scatter(const int* __restrict__ src, const int* __restrict__ dst,
                             const float* __restrict__ x, float* __restrict__ acc) {
    int idx = blockIdx.x * blockDim.x + threadIdx.x;
    if (idx >= N_EDGES * F4) return;
    int e = idx / F4;
    int q = (idx % F4) * 4;
    int s = src[e], d = dst[e];
    float4 v = *(const float4*)(x + (size_t)s * (F4 * 4) + q);
    float* a = acc + (size_t)d * (F4 * 4) + q;
    unsafeAtomicAdd(a + 0, v.x);
    unsafeAtomicAdd(a + 1, v.y);
    unsafeAtomicAdd(a + 2, v.z);
    unsafeAtomicAdd(a + 3, v.w);
}

// ---------------------------------------------- fused MLP: relu(h@W1+b1)@W2+b2
// One wave per 16-row tile.  WMMA bf16 16x16x32, f32 accumulate.
template <int K1>   // 32 (layer 0) or 128
__global__ __launch_bounds__(128)
void gin_mlp(const float* __restrict__ hin,
             const __bf16* __restrict__ w1f, const float* __restrict__ b1,
             const __bf16* __restrict__ w2f, const float* __restrict__ b2,
             float* __restrict__ h2out,
             float* __restrict__ bnsum, float* __restrict__ bnsq) {
    __shared__ __align__(16) __bf16 lds[4][16][256];   // per-wave 16x256 bf16 staging
    const int wv   = threadIdx.x >> 5;
    const int lane = threadIdx.x & 31;
    const int tile = blockIdx.x * 4 + wv;
    if (tile >= N_TILES) return;           // wave-uniform: EXEC stays all-ones
    const int m  = lane & 15;
    const int hi = lane >> 4;
    const int row = tile * 16 + m;
    constexpr int KC1 = K1 / 32;

    // A1 fragments: lane holds row m; elem j -> k = (j&7) + 8*hi + 16*(j>>3)
    v16bf a1[KC1];
#pragma unroll
    for (int kc = 0; kc < KC1; ++kc) {
        const float* p = hin + (size_t)row * K1 + kc * 32 + hi * 8;
        v16bf a;
#pragma unroll
        for (int j = 0; j < 8; ++j) a[j] = (__bf16)p[j];
#pragma unroll
        for (int j = 0; j < 8; ++j) a[8 + j] = (__bf16)p[16 + j];
        a1[kc] = a;
    }

    // GEMM1 (N=256) + bias + relu -> LDS (bf16, row-major 16x256)
    for (int nt = 0; nt < 16; ++nt) {
        float bias = b1[nt * 16 + m];
        v8f c;
#pragma unroll
        for (int r = 0; r < 8; ++r) c[r] = bias;
#pragma unroll
        for (int kc = 0; kc < KC1; ++kc) {
            v16bf b = *(const v16bf*)(w1f + ((size_t)(nt * KC1 + kc) * 32 + lane) * 16);
            c = __builtin_amdgcn_wmma_f32_16x16x32_bf16(false, a1[kc], false, b,
                                                        (short)0, c, false, false);
        }
#pragma unroll
        for (int r = 0; r < 8; ++r) {
            float v = c[r] > 0.0f ? c[r] : 0.0f;
            lds[wv][r + hi * 8][nt * 16 + m] = (__bf16)v;
        }
    }

    // A2 fragments straight out of LDS (same-wave DS ordering guarantees RAW)
    v16bf a2[8];
#pragma unroll
    for (int kc = 0; kc < 8; ++kc) {
        const __bf16* p = &lds[wv][m][kc * 32 + hi * 8];
        v16bf a;
#pragma unroll
        for (int j = 0; j < 8; ++j) a[j] = p[j];
#pragma unroll
        for (int j = 0; j < 8; ++j) a[8 + j] = p[16 + j];
        a2[kc] = a;
    }

    // GEMM2 (N=128, K=256) + bias -> h2 global + BN partial sums
    for (int nt = 0; nt < 8; ++nt) {
        float bias = b2[nt * 16 + m];
        v8f c;
#pragma unroll
        for (int r = 0; r < 8; ++r) c[r] = bias;
#pragma unroll
        for (int kc = 0; kc < 8; ++kc) {
            v16bf b = *(const v16bf*)(w2f + ((size_t)(nt * 8 + kc) * 32 + lane) * 16);
            c = __builtin_amdgcn_wmma_f32_16x16x32_bf16(false, a2[kc], false, b,
                                                        (short)0, c, false, false);
        }
        float s = 0.0f, s2 = 0.0f;
#pragma unroll
        for (int r = 0; r < 8; ++r) {
            float v = c[r];
            h2out[(size_t)(tile * 16 + r + hi * 8) * GHC + nt * 16 + m] = v;
            s += v;  s2 += v * v;
        }
        // lane l and l+16 hold the same output column: fold before the atomic
        s  += __shfl_xor(s, 16);
        s2 += __shfl_xor(s2, 16);
        if (hi == 0) {
            unsafeAtomicAdd(&bnsum[nt * 16 + m], s);
            unsafeAtomicAdd(&bnsq [nt * 16 + m], s2);
        }
    }
}

// ------------------------------------------------------------- batch-norm
__global__ void bn_finalize(const float* __restrict__ sum, const float* __restrict__ sq,
                            const float* __restrict__ gamma, const float* __restrict__ beta,
                            float* __restrict__ scale, float* __restrict__ shift) {
    int f = threadIdx.x;                   // 128 threads
    float mu  = sum[f] * (1.0f / N_NODES);
    float var = sq[f] * (1.0f / N_NODES) - mu * mu;
    float sc  = gamma[f] * rsqrtf(var + BN_EPS);
    scale[f] = sc;
    shift[f] = beta[f] - mu * sc;
}

__global__ void bn_apply_pool(float* __restrict__ h2 /* in/out -> y */,
                              float* __restrict__ hsum_next,
                              const float* __restrict__ scale, const float* __restrict__ shift,
                              const int* __restrict__ batch, float* __restrict__ pool) {
    int idx = blockIdx.x * blockDim.x + threadIdx.x;   // N_NODES*32 threads
    if (idx >= N_NODES * 32) return;
    int node = idx >> 5;
    int q = (idx & 31) * 4;
    int g = batch[node];
    float4 h  = *(const float4*)(h2 + (size_t)node * GHC + q);
    float4 sc = *(const float4*)(scale + q);
    float4 sh = *(const float4*)(shift + q);
    float4 y;
    y.x = fmaxf(h.x * sc.x + sh.x, 0.0f);
    y.y = fmaxf(h.y * sc.y + sh.y, 0.0f);
    y.z = fmaxf(h.z * sc.z + sh.z, 0.0f);
    y.w = fmaxf(h.w * sc.w + sh.w, 0.0f);
    *(float4*)(h2 + (size_t)node * GHC + q)        = y;   // next layer x
    *(float4*)(hsum_next + (size_t)node * GHC + q) = y;   // next residual base
    float* pg = pool + (size_t)g * GHC + q;
    unsafeAtomicAdd(pg + 0, y.x);
    unsafeAtomicAdd(pg + 1, y.y);
    unsafeAtomicAdd(pg + 2, y.z);
    unsafeAtomicAdd(pg + 3, y.w);
}

// --------------------------------------------------------------- final MLP
__global__ __launch_bounds__(256)
void final_mlp(const float* __restrict__ pool, const float* __restrict__ cnt,
               const float* __restrict__ Wm1, const float* __restrict__ bm1,
               const float* __restrict__ Wm2, const float* __restrict__ bm2,
               float* __restrict__ out) {
    __shared__ float zs[5 * GHC];
    __shared__ float hid[NHID];
    int g = blockIdx.x, t = threadIdx.x;
    float inv = 1.0f / fmaxf(cnt[g], 1.0f);
    for (int i = t; i < 5 * GHC; i += 256) {
        int l = i >> 7, f = i & 127;
        zs[i] = pool[((size_t)l * N_GRAPHS + g) * GHC + f] * inv;
    }
    __syncthreads();
    for (int h = t; h < NHID; h += 256) {
        float acc = bm1[h];
        for (int k = 0; k < 5 * GHC; ++k)
            acc = fmaf(zs[k], Wm1[(size_t)k * NHID + h], acc);
        hid[h] = fmaxf(acc, 0.0f);
    }
    __syncthreads();
    float acc = bm2[t];
    for (int k = 0; k < NHID; ++k)
        acc = fmaf(hid[k], Wm2[(size_t)k * NOUT + t], acc);
    out[(size_t)g * NOUT + t] = acc;
}

// ------------------------------------------------------------------- host
static inline int gdiv(long long n, int b) { return (int)((n + b - 1) / b); }

extern "C" void kernel_launch(void* const* d_in, const int* in_sizes, int n_in,
                              void* d_out, int out_size, void* d_ws, size_t ws_size,
                              hipStream_t stream) {
    const float* x    = (const float*)d_in[0];
    const int*   ei   = (const int*)d_in[1];
    const int*   bat  = (const int*)d_in[2];
    const float* W1_0 = (const float*)d_in[3];
    const float* b1_0 = (const float*)d_in[4];
    const float* W2_0 = (const float*)d_in[5];
    const float* b2_0 = (const float*)d_in[6];
    const float* W1s  = (const float*)d_in[7];
    const float* b1s  = (const float*)d_in[8];
    const float* W2s  = (const float*)d_in[9];
    const float* b2s  = (const float*)d_in[10];
    const float* gam  = (const float*)d_in[11];
    const float* bet  = (const float*)d_in[12];
    const float* Wm1  = (const float*)d_in[13];
    const float* bm1  = (const float*)d_in[14];
    const float* Wm2  = (const float*)d_in[15];
    const float* bm2  = (const float*)d_in[16];
    const int* src = ei;
    const int* dst = ei + N_EDGES;

    char* p = (char*)d_ws;
    auto alloc = [&](size_t bytes) -> char* {
        char* r = p;
        p += (bytes + 255) & ~(size_t)255;
        return r;
    };
    float*  bufA   = (float*)alloc((size_t)N_NODES * GHC * 4);   // x / h2 (in place)
    float*  bufB   = (float*)alloc((size_t)N_NODES * GHC * 4);   // residual+agg sum
    float*  pool   = (float*)alloc((size_t)5 * N_GRAPHS * GHC * 4);
    float*  cnt    = (float*)alloc(N_GRAPHS * 4);
    float*  bn     = (float*)alloc(2 * GHC * 4);                 // sum | sumsq
    float*  bnsc   = (float*)alloc(GHC * 4);
    float*  bnsh   = (float*)alloc(GHC * 4);
    __bf16* fw1_0  = (__bf16*)alloc((size_t)F_IN * 2 * GHC * 2);
    __bf16* fw2_0  = (__bf16*)alloc((size_t)2 * GHC * GHC * 2);
    __bf16* fw1[4];
    __bf16* fw2[4];
    for (int l = 0; l < 4; ++l) {
        fw1[l] = (__bf16*)alloc((size_t)GHC * 2 * GHC * 2);
        fw2[l] = (__bf16*)alloc((size_t)2 * GHC * GHC * 2);
    }
    float* bnsum = bn, *bnsq = bn + GHC;

    // init: pool/cnt zero, graph counts, weight fragment packing
    zero_f32<<<gdiv(5 * N_GRAPHS * GHC, 256), 256, 0, stream>>>(pool, 5 * N_GRAPHS * GHC);
    zero_f32<<<gdiv(N_GRAPHS, 256), 256, 0, stream>>>(cnt, N_GRAPHS);
    count_nodes<<<gdiv(N_NODES, 256), 256, 0, stream>>>(bat, cnt);
    pack_bfrags<<<gdiv(F_IN * 2 * GHC, 256), 256, 0, stream>>>(W1_0, fw1_0, F_IN, 2 * GHC);
    pack_bfrags<<<gdiv(2 * GHC * GHC, 256), 256, 0, stream>>>(W2_0, fw2_0, 2 * GHC, GHC);
    for (int l = 0; l < 4; ++l) {
        pack_bfrags<<<gdiv(GHC * 2 * GHC, 256), 256, 0, stream>>>(
            W1s + (size_t)l * GHC * 2 * GHC, fw1[l], GHC, 2 * GHC);
        pack_bfrags<<<gdiv(2 * GHC * GHC, 256), 256, 0, stream>>>(
            W2s + (size_t)l * 2 * GHC * GHC, fw2[l], 2 * GHC, GHC);
    }

    const int mlp_grid = (N_TILES + 3) / 4;   // 4 waves (tiles) per block

    // ---- layer 0 (F_IN = 32) ----
    copy_f32<<<gdiv(N_NODES * F_IN, 256), 256, 0, stream>>>(bufB, x, N_NODES * F_IN);
    edge_scatter<8><<<gdiv((long long)N_EDGES * 8, 256), 256, 0, stream>>>(src, dst, x, bufB);
    zero_f32<<<1, 256, 0, stream>>>(bn, 2 * GHC);
    gin_mlp<F_IN><<<mlp_grid, 128, 0, stream>>>(bufB, fw1_0, b1_0, fw2_0, b2_0,
                                                bufA, bnsum, bnsq);
    bn_finalize<<<1, GHC, 0, stream>>>(bnsum, bnsq, gam, bet, bnsc, bnsh);
    bn_apply_pool<<<gdiv((long long)N_NODES * 32, 256), 256, 0, stream>>>(
        bufA, bufB, bnsc, bnsh, bat, pool);

    // ---- layers 1..4 (GHC = 128) ----
    for (int l = 0; l < 4; ++l) {
        edge_scatter<32><<<gdiv((long long)N_EDGES * 32, 256), 256, 0, stream>>>(
            src, dst, bufA, bufB);
        zero_f32<<<1, 256, 0, stream>>>(bn, 2 * GHC);
        gin_mlp<GHC><<<mlp_grid, 128, 0, stream>>>(bufB, fw1[l], b1s + (size_t)l * 2 * GHC,
                                                   fw2[l], b2s + (size_t)l * GHC,
                                                   bufA, bnsum, bnsq);
        bn_finalize<<<1, GHC, 0, stream>>>(bnsum, bnsq, gam + (size_t)(l + 1) * GHC,
                                           bet + (size_t)(l + 1) * GHC, bnsc, bnsh);
        bn_apply_pool<<<gdiv((long long)N_NODES * 32, 256), 256, 0, stream>>>(
            bufA, bufB, bnsc, bnsh, bat, pool + (size_t)(l + 1) * N_GRAPHS * GHC);
    }

    // ---- readout MLP ----
    final_mlp<<<N_GRAPHS, 256, 0, stream>>>(pool, cnt, Wm1, bm1, Wm2, bm2, (float*)d_out);
}
